// MCAM_59468117180973
// MI455X (gfx1250) — matrix operand order repaired and checked by reading, not compiled, not run on hardware
//
#include <hip/hip_runtime.h>

typedef __attribute__((ext_vector_type(16))) __bf16 v16bf;
typedef __attribute__((ext_vector_type(8)))  __bf16 v8bf;
typedef __attribute__((ext_vector_type(8)))  float  v8f;

#define HW 4096
#define CD 64
#define NB 4
#define KPAD 72   // padded LDS row stride (elements): 144B, 16B-aligned, conflict-spread

union BF16x16 { v16bf v; v8bf h[2]; };

static __device__ __forceinline__ v8f wmma_bf16(v16bf a, v16bf b, v8f c) {
    return __builtin_amdgcn_wmma_f32_16x16x32_bf16(
        false, a, false, b, (short)0, c, false, false);
}

template <int M>
static __device__ __forceinline__ float shfl_xor_f(float x) {
    int i = __builtin_amdgcn_ds_swizzle(__builtin_bit_cast(int, x), (M << 10) | 0x1f);
    return __builtin_bit_cast(float, i);
}

static __device__ __forceinline__ float fexp2(float x) {
    return __builtin_amdgcn_exp2f(x);
}

// async DMA: 16 bytes per lane, global -> LDS, tracked by ASYNCcnt
static __device__ __forceinline__ void async_ld16(const __bf16* g, unsigned lds_off) {
    asm volatile("global_load_async_to_lds_b128 %0, %1, off"
                 :: "v"(lds_off), "v"((unsigned long long)(size_t)g)
                 : "memory");
}
#define ASYNC_WAIT(N) asm volatile("s_wait_asynccnt " #N ::: "memory")

static __device__ __forceinline__ void block_barrier() {
    asm volatile("" ::: "memory");
    __builtin_amdgcn_s_barrier();
    asm volatile("" ::: "memory");
}

// ---------------------------------------------------------------------------
// Projection: q/k = (N,HW,C) bf16; v stored transposed (N,C,HW).
// Q pre-scaled by log2(e) so attention works in exp2 domain.
// ---------------------------------------------------------------------------
__global__ __launch_bounds__(256) void MCAM_proj_kernel(
    const float* __restrict__ x,
    const float* __restrict__ wq, const float* __restrict__ bq,
    const float* __restrict__ wk, const float* __restrict__ bk,
    const float* __restrict__ wv, const float* __restrict__ bv,
    __bf16* __restrict__ Q, __bf16* __restrict__ K, __bf16* __restrict__ Vt)
{
    __shared__ float sw[3][64 * 64];
    __shared__ float xs[64][32];
    __shared__ float sb[3][64];

    const int t  = threadIdx.x;
    const int n  = blockIdx.y;
    const int p0 = blockIdx.x * 32;

    for (int i = t; i < 64 * 64; i += 256) {
        sw[0][i] = wq[i]; sw[1][i] = wk[i]; sw[2][i] = wv[i];
    }
    if (t < 64) { sb[0][t] = bq[t]; sb[1][t] = bk[t]; sb[2][t] = bv[t]; }
    for (int i = t; i < 64 * 32; i += 256) {
        int c = i >> 5, p = i & 31;
        xs[c][p] = x[((size_t)n * 64 + c) * HW + p0 + p];
    }
    __syncthreads();

    const int p  = t & 31;
    const int dg = t >> 5;
#pragma unroll 1
    for (int dd = 0; dd < 8; ++dd) {
        const int d = dg * 8 + dd;
        float aq = sb[0][d], ak = sb[1][d], av = sb[2][d];
#pragma unroll
        for (int c = 0; c < 64; ++c) {
            float xv = xs[c][p];
            aq = __fmaf_rn(sw[0][d * 64 + c], xv, aq);
            ak = __fmaf_rn(sw[1][d * 64 + c], xv, ak);
            av = __fmaf_rn(sw[2][d * 64 + c], xv, av);
        }
        size_t qk = ((size_t)n * HW + p0 + p) * 64 + d;
        Q[qk] = (__bf16)(aq * 1.44269504f);
        K[qk] = (__bf16)ak;
        Vt[((size_t)n * 64 + d) * HW + p0 + p] = (__bf16)av;
    }
}

// ---------------------------------------------------------------------------
// Fused dual-softmax attention (log2 domain) with async-DMA double buffering.
// ---------------------------------------------------------------------------
__global__ __launch_bounds__(128) void MCAM_attn_kernel(
    const __bf16* __restrict__ Qo, const __bf16* __restrict__ Ko, const __bf16* __restrict__ Vo,
    const __bf16* __restrict__ Qs, const __bf16* __restrict__ Ks, const __bf16* __restrict__ Vs,
    float* __restrict__ out)
{
    __shared__ __bf16 Kst[2][2][64 * KPAD];   // [buf][mod] 36 KB
    __shared__ __bf16 Vst[2][64 * KPAD];      // [mod]      18 KB
    __shared__ __bf16 Pbuf[4][16][64];        //             8 KB

    const int t    = threadIdx.x;
    const int lane = t & 31;
    const int wave = t >> 5;
    const int n    = blockIdx.y;
    const int q0   = blockIdx.x * 64 + wave * 16;
    const int col  = lane & 15;
    const int hi   = lane >> 4;

    // staging mapping: 128 threads x 16B cover 16 rows x 128B per statement
    const int srow  = t >> 3;          // row within 16-row chunk
    const int scolE = (t & 7) * 8;     // element offset within row
    const int scolB = (t & 7) * 16;    // byte offset within LDS row

    const unsigned LK00 = (unsigned)(size_t)&Kst[0][0][0];
    const unsigned LK01 = (unsigned)(size_t)&Kst[0][1][0];
    const unsigned LK10 = (unsigned)(size_t)&Kst[1][0][0];
    const unsigned LK11 = (unsigned)(size_t)&Kst[1][1][0];
    const unsigned LV0  = (unsigned)(size_t)&Vst[0][0];
    const unsigned LV1  = (unsigned)(size_t)&Vst[1][0];

    auto stage = [&](const __bf16* g, int gstride, unsigned lbase) {
#pragma unroll
        for (int j = 0; j < 4; ++j) {
            int r = j * 16 + srow;
            async_ld16(g + (size_t)r * gstride + scolE,
                       lbase + (unsigned)(r * (KPAD * 2) ) + scolB);
        }
    };

    // --- Q A-fragments (registers for whole kernel) ---
    BF16x16 qa_o[2], qa_s[2];
    {
        const __bf16* qro = Qo + ((size_t)n * HW + q0 + col) * 64;
        const __bf16* qrs = Qs + ((size_t)n * HW + q0 + col) * 64;
#pragma unroll
        for (int cc = 0; cc < 2; ++cc) {
            qa_o[cc].h[0] = *(const v8bf*)(qro + cc * 32 + hi * 8);
            qa_o[cc].h[1] = *(const v8bf*)(qro + cc * 32 + 16 + hi * 8);
            qa_s[cc].h[0] = *(const v8bf*)(qrs + cc * 32 + hi * 8);
            qa_s[cc].h[1] = *(const v8bf*)(qrs + cc * 32 + 16 + hi * 8);
        }
    }

    const __bf16* kbo = Ko + (size_t)n * HW * 64;
    const __bf16* kbs = Ks + (size_t)n * HW * 64;
    const __bf16* vbo = Vo + (size_t)n * 64 * HW;
    const __bf16* vbs = Vs + (size_t)n * 64 * HW;

    // prologue: start K tile 0 into buffer 0
    stage(kbo, 64, LK00);
    stage(kbs, 64, LK01);

    v8f acc_o[4], acc_s[4];
#pragma unroll
    for (int i = 0; i < 4; ++i) { acc_o[i] = v8f{}; acc_s[i] = v8f{}; }
    float m_o[8], l_o[8], m_s[8], l_s[8];
#pragma unroll
    for (int r = 0; r < 8; ++r) {
        m_o[r] = -3.0e38f; m_s[r] = -3.0e38f; l_o[r] = 0.0f; l_s[r] = 0.0f;
    }

    for (int kt = 0; kt < 64; ++kt) {
        const int par = kt & 1;

        // K[kt] (our 8 copies) landed; everyone else's too after the barrier,
        // and all waves are done reading V[kt-1] / K[kt-1].
        ASYNC_WAIT(0);
        block_barrier();

        // issue V[kt] first (8), then K[kt+1] (8): in-order ASYNCcnt lets us
        // release V with "<=8" later while K keeps flying.
        stage(vbo + kt * 64, HW, LV0);
        stage(vbs + kt * 64, HW, LV1);
        const int ktn = (kt + 1) & 63;
        stage(kbo + (size_t)ktn * 4096, 64, par ? LK00 : LK10);
        stage(kbs + (size_t)ktn * 4096, 64, par ? LK01 : LK11);

        const __bf16* kp_o = &Kst[par][0][0];
        const __bf16* kp_s = &Kst[par][1][0];

        // ---- S = Q K^T from LDS ----
        v8f S_o[4], S_s[4];
#pragma unroll
        for (int st = 0; st < 4; ++st) {
            const int rb = (st * 16 + col) * KPAD + hi * 8;
            BF16x16 kb;
            v8f s = v8f{};
            kb.h[0] = *(const v8bf*)(kp_o + rb);
            kb.h[1] = *(const v8bf*)(kp_o + rb + 16);
            s = wmma_bf16(qa_o[0].v, kb.v, s);
            kb.h[0] = *(const v8bf*)(kp_o + rb + 32);
            kb.h[1] = *(const v8bf*)(kp_o + rb + 48);
            s = wmma_bf16(qa_o[1].v, kb.v, s);
            S_o[st] = s;

            s = v8f{};
            kb.h[0] = *(const v8bf*)(kp_s + rb);
            kb.h[1] = *(const v8bf*)(kp_s + rb + 16);
            s = wmma_bf16(qa_s[0].v, kb.v, s);
            kb.h[0] = *(const v8bf*)(kp_s + rb + 32);
            kb.h[1] = *(const v8bf*)(kp_s + rb + 48);
            s = wmma_bf16(qa_s[1].v, kb.v, s);
            S_s[st] = s;
        }

        // ---- per-row tile maxes ----
        float cm_o[8], cm_s[8];
#pragma unroll
        for (int r = 0; r < 8; ++r) {
            float a = fmaxf(fmaxf(S_o[0][r], S_o[1][r]), fmaxf(S_o[2][r], S_o[3][r]));
            float b = fmaxf(fmaxf(S_s[0][r], S_s[1][r]), fmaxf(S_s[2][r], S_s[3][r]));
            a = fmaxf(a, shfl_xor_f<1>(a)); a = fmaxf(a, shfl_xor_f<2>(a));
            a = fmaxf(a, shfl_xor_f<4>(a)); a = fmaxf(a, shfl_xor_f<8>(a));
            b = fmaxf(b, shfl_xor_f<1>(b)); b = fmaxf(b, shfl_xor_f<2>(b));
            b = fmaxf(b, shfl_xor_f<4>(b)); b = fmaxf(b, shfl_xor_f<8>(b));
            cm_o[r] = a; cm_s[r] = b;
        }

        // ---- online rescale (log2 domain) ----
#pragma unroll
        for (int r = 0; r < 8; ++r) {
            float mo_n = fmaxf(m_o[r], cm_o[r]);
            float ms_n = fmaxf(m_s[r], cm_s[r]);
            float sc   = fexp2((m_o[r] - mo_n) + (m_s[r] - ms_n));
            l_o[r] *= fexp2(m_o[r] - mo_n);
            l_s[r] *= fexp2(m_s[r] - ms_n);
#pragma unroll
            for (int ct = 0; ct < 4; ++ct) { acc_o[ct][r] *= sc; acc_s[ct][r] *= sc; }
            m_o[r] = mo_n; m_s[r] = ms_n;
        }

        // ---- P = exp2(So-mo)*exp2(Ss-ms) -> LDS; per-lane l partials ----
#pragma unroll
        for (int st = 0; st < 4; ++st) {
#pragma unroll
            for (int r = 0; r < 8; ++r) {
                float eo = fexp2(S_o[st][r] - m_o[r]);
                float es = fexp2(S_s[st][r] - m_s[r]);
                l_o[r] += eo; l_s[r] += es;
                Pbuf[wave][r + hi * 8][st * 16 + col] = (__bf16)(eo * es);
            }
        }

        // V[kt] (first 8 issues) arrived; K[kt+1] may still be in flight.
        ASYNC_WAIT(8);
        block_barrier();

        // ---- acc += P @ V ----
        BF16x16 pa[2];
        const __bf16* prow = &Pbuf[wave][col][0];
#pragma unroll
        for (int kc = 0; kc < 2; ++kc) {
            pa[kc].h[0] = *(const v8bf*)(prow + kc * 32 + hi * 8);
            pa[kc].h[1] = *(const v8bf*)(prow + kc * 32 + 16 + hi * 8);
        }
#pragma unroll
        for (int ct = 0; ct < 4; ++ct) {
            const int rb = (ct * 16 + col) * KPAD + hi * 8;
#pragma unroll
            for (int kc = 0; kc < 2; ++kc) {
                BF16x16 vb;
                vb.h[0] = *(const v8bf*)(&Vst[0][0] + rb + kc * 32);
                vb.h[1] = *(const v8bf*)(&Vst[0][0] + rb + kc * 32 + 16);
                acc_o[ct] = wmma_bf16(pa[kc].v, vb.v, acc_o[ct]);
                vb.h[0] = *(const v8bf*)(&Vst[1][0] + rb + kc * 32);
                vb.h[1] = *(const v8bf*)(&Vst[1][0] + rb + kc * 32 + 16);
                acc_s[ct] = wmma_bf16(pa[kc].v, vb.v, acc_s[ct]);
            }
        }
    }

    // ---- epilogue: reduce l partials once, normalize, multiply ----
#pragma unroll
    for (int r = 0; r < 8; ++r) {
        float a = l_o[r];
        a += shfl_xor_f<1>(a); a += shfl_xor_f<2>(a);
        a += shfl_xor_f<4>(a); a += shfl_xor_f<8>(a);
        float b = l_s[r];
        b += shfl_xor_f<1>(b); b += shfl_xor_f<2>(b);
        b += shfl_xor_f<4>(b); b += shfl_xor_f<8>(b);
        float inv = 1.0f / (a * b);
        int q = q0 + r + hi * 8;
#pragma unroll
        for (int ct = 0; ct < 4; ++ct) {
            int c = ct * 16 + col;
            float vo = acc_o[ct][r] * inv;
            float vs = acc_s[ct][r] * inv;
            out[((size_t)n * 64 + c) * HW + q] = vo * vs;
        }
    }
}

extern "C" void kernel_launch(void* const* d_in, const int* in_sizes, int n_in,
                              void* d_out, int out_size, void* d_ws, size_t ws_size,
                              hipStream_t stream) {
    (void)in_sizes; (void)n_in; (void)out_size; (void)ws_size;
    const float* x_opt = (const float*)d_in[0];
    const float* x_sar = (const float*)d_in[1];
    const float* wq_o = (const float*)d_in[2];   const float* bq_o = (const float*)d_in[3];
    const float* wk_o = (const float*)d_in[4];   const float* bk_o = (const float*)d_in[5];
    const float* wv_o = (const float*)d_in[6];   const float* bv_o = (const float*)d_in[7];
    const float* wq_s = (const float*)d_in[8];   const float* bq_s = (const float*)d_in[9];
    const float* wk_s = (const float*)d_in[10];  const float* bk_s = (const float*)d_in[11];
    const float* wv_s = (const float*)d_in[12];  const float* bv_s = (const float*)d_in[13];

    const size_t elems = (size_t)NB * HW * CD;
    __bf16* Qo = (__bf16*)d_ws;
    __bf16* Ko = Qo + elems;
    __bf16* Vo = Ko + elems;
    __bf16* Qs = Vo + elems;
    __bf16* Ks = Qs + elems;
    __bf16* Vs = Ks + elems;

    dim3 pgrid(HW / 32, NB);
    MCAM_proj_kernel<<<pgrid, 256, 0, stream>>>(x_opt, wq_o, bq_o, wk_o, bk_o, wv_o, bv_o, Qo, Ko, Vo);
    MCAM_proj_kernel<<<pgrid, 256, 0, stream>>>(x_sar, wq_s, bq_s, wk_s, bk_s, wv_s, bv_s, Qs, Ks, Vs);

    dim3 agrid(HW / 64, NB);
    MCAM_attn_kernel<<<agrid, 128, 0, stream>>>(Qo, Ko, Vo, Qs, Ks, Vs, (float*)d_out);
}
